// self_attn_20349555048454
// MI455X (gfx1250) — compile-verified, matrix operand order checked by reading, not run on hardware
//
#include <hip/hip_runtime.h>

// Problem constants (match reference): B=4, C=256, C8=32, H=W=64, N=4096
#define BB 4
#define CC 256
#define C8 32
#define NN 4096

typedef __attribute__((ext_vector_type(16))) __bf16 v16bf;
typedef __attribute__((ext_vector_type(8)))  float  v8f;

#if __has_builtin(__builtin_amdgcn_global_load_async_to_lds_b128)
#define USE_ASYNC_LDS 1
typedef int v4i __attribute__((vector_size(16)));           // generic vector int4
typedef __attribute__((address_space(1))) v4i gv4i;         // global
typedef __attribute__((address_space(3))) v4i lv4i;         // LDS
#endif

union Frag {
  v16bf bf;
  uint4 q[2];
};

__device__ __forceinline__ unsigned short f2bf(float f) {
  unsigned int u = __builtin_bit_cast(unsigned int, f);
  u += 0x7FFFu + ((u >> 16) & 1u);   // round-to-nearest-even
  return (unsigned short)(u >> 16);
}

// 16-bit A/B WMMA fragment: lane's K-runs are [8h, 8h+8) and [16+8h, 24+8h)
__device__ __forceinline__ Frag load_frag(const unsigned short* row, int half) {
  Frag f;
  f.q[0] = *reinterpret_cast<const uint4*>(row + 8 * half);
  f.q[1] = *reinterpret_cast<const uint4*>(row + 16 + 8 * half);
  return f;
}

__device__ __forceinline__ v8f wmma_bf16(const Frag& a, const Frag& b, v8f c) {
  return __builtin_amdgcn_wmma_f32_16x16x32_bf16(false, a.bf, false, b.bf,
                                                 (short)0, c, false, false);
}

__device__ __forceinline__ void wait_async_zero() {
#ifdef USE_ASYNC_LDS
#if __has_builtin(__builtin_amdgcn_s_wait_asynccnt)
  __builtin_amdgcn_s_wait_asynccnt(0);
#else
  asm volatile("s_wait_asynccnt 0x0" ::: "memory");
#endif
#endif
}

// ---------------------------------------------------------------------------
// Kernel A: x [B,C,N] f32  ->  xT [B,N,C] bf16   (tiled 64x64 transpose)
// ---------------------------------------------------------------------------
__global__ __launch_bounds__(256) void transpose_convert(
    const float* __restrict__ x, unsigned short* __restrict__ xT) {
  __shared__ float tile[64][65];
  const int b  = blockIdx.z;
  const int n0 = blockIdx.y * 64;
  const int c0 = blockIdx.x * 64;
  const int j  = threadIdx.x;                  // n within tile
  for (int i = threadIdx.y; i < 64; i += 4)
    tile[i][j] = x[((size_t)(b * CC + c0 + i)) * NN + n0 + j];
  __syncthreads();
  const int ci = threadIdx.x;                  // c within tile
  for (int r = threadIdx.y; r < 64; r += 4)
    xT[((size_t)(b * NN + n0 + r)) * CC + c0 + ci] = f2bf(tile[ci][r]);
}

// ---------------------------------------------------------------------------
// Kernel B: convert Wq(32x256) | Wk(32x256) | Wv(256x256) f32 -> bf16
// ---------------------------------------------------------------------------
__global__ __launch_bounds__(256) void convert_weights(
    const float* __restrict__ Wq, const float* __restrict__ Wk,
    const float* __restrict__ Wv, unsigned short* __restrict__ dst) {
  int i = blockIdx.x * 256 + threadIdx.x;      // 0 .. 81919
  float f;
  if (i < 8192)        f = Wq[i];
  else if (i < 16384)  f = Wk[i - 8192];
  else                 f = Wv[i - 16384];
  dst[i] = f2bf(f);
}

// ---------------------------------------------------------------------------
// Kernel C: projections via WMMA, 2-deep double-buffered B-fragment stream.
//   out[n][o] = sum_c xT[n][c] * W[o][c] + bias[o]
//   fragment stream f = 0..19 : q(2) | k(2) | v(16)
// ---------------------------------------------------------------------------
__device__ __forceinline__ const unsigned short* w_row(
    const unsigned short* wq, const unsigned short* wk,
    const unsigned short* wv, int f, int col) {
  if (f < 2) return wq + (size_t)(f * 16 + col) * CC;
  if (f < 4) return wk + (size_t)((f - 2) * 16 + col) * CC;
  return wv + (size_t)((f - 4) * 16 + col) * CC;
}

__global__ __launch_bounds__(128) void proj_kernel(
    const unsigned short* __restrict__ xT, const unsigned short* __restrict__ wbuf,
    const float* __restrict__ bq, const float* __restrict__ bk,
    const float* __restrict__ bv,
    unsigned short* __restrict__ qb, unsigned short* __restrict__ ktb,
    unsigned short* __restrict__ vb) {
  const int b    = blockIdx.y;
  const int wave = threadIdx.x >> 5;
  const int lane = threadIdx.x & 31;
  const int col  = lane & 15;
  const int half = lane >> 4;
  const int nbase = blockIdx.x * 64 + wave * 16;

  const unsigned short* wq = wbuf;
  const unsigned short* wk = wbuf + 32 * 256;
  const unsigned short* wv = wbuf + 64 * 256;

  v8f zero = {0.f, 0.f, 0.f, 0.f, 0.f, 0.f, 0.f, 0.f};
  v8f acc[20];
#pragma unroll
  for (int f = 0; f < 20; ++f) acc[f] = zero;

  const unsigned short* xrow = xT + ((size_t)(b * NN + nbase + col)) * CC;

  Frag acur = load_frag(xrow, half);
  // Two B-fragment buffers with overlapping live ranges -> no coalescing,
  // loads issue ~2 WMMAs ahead of their consumer.
  Frag b0 = load_frag(w_row(wq, wk, wv, 0, col), half);
  Frag b1 = load_frag(w_row(wq, wk, wv, 1, col), half);

#pragma unroll 1
  for (int kc = 0; kc < CC; kc += 32) {
    const int kcn = (kc + 32 < CC) ? kc + 32 : kc;
    Frag anext = load_frag(xrow + kcn, half);
#pragma unroll
    for (int f = 0; f < 20; f += 2) {
      const int f2  = f + 2;
      const int pf0 = (f2 < 20) ? f2 : 0;
      const int pk0 = (f2 < 20) ? kc : kcn;
      const int f3  = f + 3;
      const int pf1 = (f3 < 20) ? f3 - ((f3 >= 20) ? 20 : 0) : f3;  // equals f3 on both branches; unused
      const int pk1 = (f3 < 20) ? kc : kcn;
      acc[f] = wmma_bf16(acur, b0, acc[f]);
      b0 = load_frag(w_row(wq, wk, wv, pf0, col) + pk0, half);
      acc[f + 1] = wmma_bf16(acur, b1, acc[f + 1]);
      b1 = load_frag(w_row(wq, wk, wv, (f3 < 20) ? f3 : f3 - 20, col) + pk1, half);
    }
    acur = anext;
  }

  // D layout: lane = column o, VGPR i = row (i + 8*half)
#pragma unroll
  for (int t = 0; t < 2; ++t) {
    int o = t * 16 + col;
    float biasq = bq[o], biask = bk[o];
#pragma unroll
    for (int i = 0; i < 8; ++i) {
      int n = nbase + 8 * half + i;
      qb [(size_t)(b * NN + n) * C8 + o] = f2bf(acc[t][i] + biasq);
      ktb[(size_t)(b * NN + n) * C8 + o] = f2bf(acc[2 + t][i] + biask);
    }
  }
#pragma unroll
  for (int j = 0; j < 16; ++j) {
    int c = j * 16 + col;
    float bias = bv[c];
    unsigned int p[4];
#pragma unroll
    for (int i = 0; i < 4; ++i) {
      unsigned int lo = f2bf(acc[4 + j][2 * i] + bias);
      unsigned int hi = f2bf(acc[4 + j][2 * i + 1] + bias);
      p[i] = lo | (hi << 16);
    }
    uint4 st; st.x = p[0]; st.y = p[1]; st.z = p[2]; st.w = p[3];
    *reinterpret_cast<uint4*>(vb + ((size_t)(b * CC + c)) * NN + nbase + 8 * half) = st;
  }
}

// ---------------------------------------------------------------------------
// Kernel D: fused sigmoid-attention, double-buffered async V staging.
// ---------------------------------------------------------------------------
#define VSTRIDE 40   // ushorts per LDS row (80B, 16B-aligned, conflict-padded)
#define NITER   (NN / 32)

// Stage V tile [256 c][32 m] into LDS buffer; each thread copies 2 rows (64B).
__device__ __forceinline__ void stage_vtile(unsigned short* vt,
                                            const unsigned short* vb_b,
                                            int m0, int tid) {
#pragma unroll
  for (int r = 0; r < 2; ++r) {
    int c = tid * 2 + r;
    const unsigned short* src = vb_b + (size_t)c * NN + m0;
    unsigned short* dst = vt + c * VSTRIDE;
#ifdef USE_ASYNC_LDS
    // IOFFSET applies to both global and LDS addresses -> literal immediates
    __builtin_amdgcn_global_load_async_to_lds_b128((gv4i*)(src), (lv4i*)(dst), 0, 0);
    __builtin_amdgcn_global_load_async_to_lds_b128((gv4i*)(src), (lv4i*)(dst), 16, 0);
    __builtin_amdgcn_global_load_async_to_lds_b128((gv4i*)(src), (lv4i*)(dst), 32, 0);
    __builtin_amdgcn_global_load_async_to_lds_b128((gv4i*)(src), (lv4i*)(dst), 48, 0);
#else
#pragma unroll
    for (int k2 = 0; k2 < 4; ++k2)
      reinterpret_cast<uint4*>(dst)[k2] =
          reinterpret_cast<const uint4*>(src)[k2];
#endif
  }
}

__global__ __launch_bounds__(128) void attn_kernel(
    const unsigned short* __restrict__ qb, const unsigned short* __restrict__ ktb,
    const unsigned short* __restrict__ vb, const float* __restrict__ x,
    const float* __restrict__ gamma_p, float* __restrict__ out) {
  __shared__ unsigned short vtile[2][CC * VSTRIDE];   // 2 x 20 KB
  __shared__ unsigned short ptile[4 * 16 * VSTRIDE];  // per-wave P, 5 KB

  const int b    = blockIdx.y;
  const int tid  = threadIdx.x;
  const int wave = tid >> 5;
  const int lane = tid & 31;
  const int col  = lane & 15;
  const int half = lane >> 4;
  const int nbase = blockIdx.x * 64 + wave * 16;

  const unsigned short* vb_b  = vb + (size_t)b * CC * NN;
  const unsigned short* ktb_b = ktb + (size_t)b * NN * C8;

  Frag qf = load_frag(qb + (size_t)(b * NN + nbase + col) * C8, half);

  v8f zero = {0.f, 0.f, 0.f, 0.f, 0.f, 0.f, 0.f, 0.f};
  v8f acc[16];
#pragma unroll
  for (int j = 0; j < 16; ++j) acc[j] = zero;

  const float NLOG2E = -1.4426950408889634f;

  // Prologue: stage tile 0, prefetch K fragments for tile 0.
  stage_vtile(vtile[0], vb_b, 0, tid);
  Frag kf0 = load_frag(ktb_b + (size_t)(col) * C8, half);
  Frag kf1 = load_frag(ktb_b + (size_t)(16 + col) * C8, half);

#pragma unroll 1
  for (int mi = 0; mi < NITER; ++mi) {
    const int cur = mi & 1;
    const int m0  = mi * 32;
    wait_async_zero();     // my async loads into vtile[cur] complete
    __syncthreads();       // everyone's staging done; prev compute done

#ifdef USE_ASYNC_LDS
    if (mi + 1 < NITER) stage_vtile(vtile[1 - cur], vb_b, m0 + 32, tid);
#endif

    // Prefetch next iteration's K fragments (overlapping live ranges with
    // kf0/kf1 -> distinct registers -> loads issue a full iteration early)
    const int mn = (mi + 1 < NITER) ? m0 + 32 : m0;
    Frag nk0 = load_frag(ktb_b + (size_t)(mn + col) * C8, half);
    Frag nk1 = load_frag(ktb_b + (size_t)(mn + 16 + col) * C8, half);

    // S = Q x K (two 16-m sub-tiles) -> sigmoid -> bf16 P scatter into LDS
    v8f s0 = wmma_bf16(qf, kf0, zero);
    v8f s1 = wmma_bf16(qf, kf1, zero);
#pragma unroll
    for (int i = 0; i < 8; ++i) {
      float e0 = __builtin_amdgcn_exp2f(NLOG2E * s0[i]);
      float e1 = __builtin_amdgcn_exp2f(NLOG2E * s1[i]);
      unsigned short p0 = f2bf(__builtin_amdgcn_rcpf(1.0f + e0));
      unsigned short p1 = f2bf(__builtin_amdgcn_rcpf(1.0f + e1));
      ptile[(wave * 16 + 8 * half + i) * VSTRIDE + col]      = p0;
      ptile[(wave * 16 + 8 * half + i) * VSTRIDE + 16 + col] = p1;
    }
    kf0 = nk0;
    kf1 = nk1;

    // D->A transpose readback (DS ops are in-order within a wave)
    Frag pf;
    const unsigned short* prow = &ptile[(wave * 16 + col) * VSTRIDE];
    pf.q[0] = *reinterpret_cast<const uint4*>(prow + 8 * half);
    pf.q[1] = *reinterpret_cast<const uint4*>(prow + 16 + 8 * half);

    // O += P x V : 16 channel slabs, 2-deep double-buffered LDS fragments
    const unsigned short* vt = vtile[cur];
    auto ldsv = [&](int j) {
      Frag f;
      const unsigned short* vrow = vt + (size_t)(j * 16 + col) * VSTRIDE;
      f.q[0] = *reinterpret_cast<const uint4*>(vrow + 8 * half);
      f.q[1] = *reinterpret_cast<const uint4*>(vrow + 16 + 8 * half);
      return f;
    };
    Frag vf0 = ldsv(0);
    Frag vf1 = ldsv(1);
#pragma unroll
    for (int j = 0; j < 16; j += 2) {
      acc[j] = wmma_bf16(pf, vf0, acc[j]);
      if (j + 2 < 16) vf0 = ldsv(j + 2);
      acc[j + 1] = wmma_bf16(pf, vf1, acc[j + 1]);
      if (j + 3 < 16) vf1 = ldsv(j + 3);
    }

#ifndef USE_ASYNC_LDS
    if (mi + 1 < NITER) stage_vtile(vtile[1 - cur], vb_b, m0 + 32, tid);
#endif
  }

  // Epilogue: out = gamma*O + x ; lane = channel col, rows = 8 consecutive n
  float gamma = gamma_p[0];
#pragma unroll
  for (int j = 0; j < 16; ++j) {
    int c = j * 16 + col;
    size_t base = ((size_t)(b * CC + c)) * NN + nbase + 8 * half;
    float4 x0 = *reinterpret_cast<const float4*>(x + base);
    float4 x1 = *reinterpret_cast<const float4*>(x + base + 4);
    float4 o0, o1;
    o0.x = gamma * acc[j][0] + x0.x;  o0.y = gamma * acc[j][1] + x0.y;
    o0.z = gamma * acc[j][2] + x0.z;  o0.w = gamma * acc[j][3] + x0.w;
    o1.x = gamma * acc[j][4] + x1.x;  o1.y = gamma * acc[j][5] + x1.y;
    o1.z = gamma * acc[j][6] + x1.z;  o1.w = gamma * acc[j][7] + x1.w;
    *reinterpret_cast<float4*>(out + base)     = o0;
    *reinterpret_cast<float4*>(out + base + 4) = o1;
  }
}

// ---------------------------------------------------------------------------
extern "C" void kernel_launch(void* const* d_in, const int* in_sizes, int n_in,
                              void* d_out, int out_size, void* d_ws, size_t ws_size,
                              hipStream_t stream) {
  const float* x     = (const float*)d_in[0];
  const float* Wq    = (const float*)d_in[1];
  const float* bq    = (const float*)d_in[2];
  const float* Wk    = (const float*)d_in[3];
  const float* bk    = (const float*)d_in[4];
  const float* Wv    = (const float*)d_in[5];
  const float* bv    = (const float*)d_in[6];
  const float* gamma = (const float*)d_in[7];
  float* out = (float*)d_out;

  char* ws = (char*)d_ws;
  // ws layout (bytes):
  //   [0,1M)    q_bf  [B][N][32]
  //   [1M,2M)   kT_bf [B][N][32]
  //   [2M,10M)  v_bf  [B][C][N]
  //   [10M,18M) xT_bf [B][N][C]
  //   [18M,..)  W bf16: wq(8192) | wk(8192) | wv(65536)
  unsigned short* qb   = (unsigned short*)(ws);
  unsigned short* ktb  = (unsigned short*)(ws + (size_t)(1u << 20));
  unsigned short* vb   = (unsigned short*)(ws + (size_t)(2u << 20));
  unsigned short* xT   = (unsigned short*)(ws + (size_t)(10u << 20));
  unsigned short* wbuf = (unsigned short*)(ws + (size_t)(18u << 20));

  transpose_convert<<<dim3(4, 64, 4), dim3(64, 4), 0, stream>>>(x, xT);
  convert_weights<<<dim3(320), dim3(256), 0, stream>>>(Wq, Wk, Wv, wbuf);
  proj_kernel<<<dim3(64, 4), dim3(128), 0, stream>>>(xT, wbuf, bq, bk, bv,
                                                     qb, ktb, vb);
  attn_kernel<<<dim3(64, 4), dim3(128), 0, stream>>>(qb, ktb, vb, x, gamma, out);
}
